// Fusion_73435350827602
// MI455X (gfx1250) — matrix operand order repaired and checked by reading, not compiled, not run on hardware
//
#include <hip/hip_runtime.h>
#include <math.h>

#define DEVINL __device__ __forceinline__

typedef __attribute__((ext_vector_type(16))) __bf16 bf16x16;
typedef __attribute__((ext_vector_type(8)))  float  f32x8;

struct alignas(16) V4 { unsigned x, y, z, w; };

DEVINL __bf16 f2bf(float f) {
  union { float f; unsigned u; } v; v.f = f;
  unsigned r = v.u + 0x7FFFu + ((v.u >> 16) & 1u);
  union { unsigned short s; __bf16 b; } o; o.s = (unsigned short)(r >> 16);
  return o.b;
}
DEVINL float bf2f(__bf16 b) {
  union { unsigned short s; __bf16 b; } i; i.b = b;
  union { unsigned u; float f; } v; v.u = ((unsigned)i.s) << 16;
  return v.f;
}

// ---------------------------------------------------------------------------
// Tiled bf16 GEMM: C[M,N] = epilogue( alpha * A[M,K] x B[K,N] )
//  TA: A given transposed (element (m,k) at A[k*lda+m])
//  TB: B given transposed (element (k,n) at B[n*ldb+k])
//  ROWSS: v = v*rowScale[m]+rowShift[m];  COLB: v += colBias[n];  RELU: max(v,0)
//  WF32 / WBF: which output(s) to write (row-major, ldc)
//  Block: 256 threads = 8 waves; macro tile 128x128; K-step 32.
//  Double-buffered LDS pipeline: next tile's global loads are issued before
//  the WMMA block so vmem latency overlaps matrix math; one barrier per step.
// ---------------------------------------------------------------------------
template<bool TA, bool TB, bool ROWSS, bool COLB, bool RELU, bool WF32, bool WBF>
__global__ __launch_bounds__(256)
void gemm_bf16(const __bf16* __restrict__ A, const __bf16* __restrict__ Bm,
               float* __restrict__ Cf, __bf16* __restrict__ Cb,
               int M, int N, int K, int lda, int ldb, int ldc,
               long sAz, long sBz, long sCz,
               const float* __restrict__ rowScale, const float* __restrict__ rowShift,
               const float* __restrict__ colBias, float alpha) {
  (void)M; (void)N;
  int z = blockIdx.z;
  A  += (long)z * sAz;
  Bm += (long)z * sBz;
  if (WF32) Cf += (long)z * sCz;
  if (WBF)  Cb += (long)z * sCz;

  const int m0 = blockIdx.y * 128;
  const int n0 = blockIdx.x * 128;
  const int tid  = threadIdx.x;
  const int lane = tid & 31;
  const int wave = tid >> 5;
  const int wm = wave >> 2;     // 0..1 : 64-row slab
  const int wn = wave & 3;      // 0..3 : 32-col slab
  const int row = lane & 15;
  const int hi  = lane >> 4;

  __shared__ __bf16 sA[2][128 * 32];   // [m][k], ping-pong
  __shared__ __bf16 sB[2][128 * 32];   // [n][k], ping-pong

  f32x8 acc[4][2];
  const f32x8 zero = {0.f, 0.f, 0.f, 0.f, 0.f, 0.f, 0.f, 0.f};
#pragma unroll
  for (int mt = 0; mt < 4; mt++)
#pragma unroll
    for (int nt = 0; nt < 2; nt++) acc[mt][nt] = zero;

  // ---- register staging for global -> LDS ----
  V4 ra0, ra1, rb0, rb1;
  auto load_tileA = [&](int k0) {
    if (!TA) {
      const __bf16* p = A + (long)(m0 + (tid >> 1)) * lda + k0 + (tid & 1) * 16;
      ra0 = ((const V4*)p)[0]; ra1 = ((const V4*)p)[1];
      if (k0 + 32 < K) __builtin_prefetch(p + 32, 0, 3);
    } else {
      const __bf16* p = A + (long)(k0 + (tid >> 3)) * lda + m0 + (tid & 7) * 16;
      ra0 = ((const V4*)p)[0]; ra1 = ((const V4*)p)[1];
    }
  };
  auto store_tileA = [&](__bf16* s) {
    if (!TA) {
      V4* q = (V4*)(s + (tid >> 1) * 32 + (tid & 1) * 16);
      q[0] = ra0; q[1] = ra1;
    } else {
      union { V4 q[2]; __bf16 h[16]; } u; u.q[0] = ra0; u.q[1] = ra1;
      int kk = tid >> 3, mh = (tid & 7) * 16;
#pragma unroll
      for (int e = 0; e < 16; e++) s[(mh + e) * 32 + kk] = u.h[e];
    }
  };
  auto load_tileB = [&](int k0) {
    if (TB) {
      const __bf16* p = Bm + (long)(n0 + (tid >> 1)) * ldb + k0 + (tid & 1) * 16;
      rb0 = ((const V4*)p)[0]; rb1 = ((const V4*)p)[1];
      if (k0 + 32 < K) __builtin_prefetch(p + 32, 0, 3);
    } else {
      const __bf16* p = Bm + (long)(k0 + (tid >> 3)) * ldb + n0 + (tid & 7) * 16;
      rb0 = ((const V4*)p)[0]; rb1 = ((const V4*)p)[1];
    }
  };
  auto store_tileB = [&](__bf16* s) {
    if (TB) {
      V4* q = (V4*)(s + (tid >> 1) * 32 + (tid & 1) * 16);
      q[0] = rb0; q[1] = rb1;
    } else {
      union { V4 q[2]; __bf16 h[16]; } u; u.q[0] = rb0; u.q[1] = rb1;
      int kk = tid >> 3, nh = (tid & 7) * 16;
#pragma unroll
      for (int e = 0; e < 16; e++) s[(nh + e) * 32 + kk] = u.h[e];
    }
  };

  // ---- prologue: stage first tile ----
  load_tileA(0);
  load_tileB(0);
  store_tileA(sA[0]);
  store_tileB(sB[0]);
  __syncthreads();

  int cur = 0;
  for (int k0 = 0; k0 < K; k0 += 32) {
    const bool more = (k0 + 32) < K;
    // issue next tile's global loads now; they complete while WMMAs run
    if (more) { load_tileA(k0 + 32); load_tileB(k0 + 32); }

    // ---- gather WMMA fragments from current LDS buffer ----
    const __bf16* __restrict__ cA = sA[cur];
    const __bf16* __restrict__ cB = sB[cur];
    bf16x16 af[4], bfr[2];
#pragma unroll
    for (int mt = 0; mt < 4; mt++) {
      int m = wm * 64 + mt * 16 + row;
      const V4* pa = (const V4*)(cA + m * 32);
      union { V4 q[2]; bf16x16 v; } ua;
      ua.q[0] = pa[2 * hi];       // K = hi*16 + 0..7
      ua.q[1] = pa[2 * hi + 1];   // K = hi*16 + 8..15
      af[mt] = ua.v;
    }
#pragma unroll
    for (int nt = 0; nt < 2; nt++) {
      int n = wn * 32 + nt * 16 + row;
      const V4* pb = (const V4*)(cB + n * 32 + hi * 16);
      union { V4 q[2]; bf16x16 v; } ub;
      ub.q[0] = pb[0]; ub.q[1] = pb[1];   // K = hi*16 + 0..15
      bfr[nt] = ub.v;
    }
#pragma unroll
    for (int mt = 0; mt < 4; mt++)
#pragma unroll
      for (int nt = 0; nt < 2; nt++)
        acc[mt][nt] = __builtin_amdgcn_wmma_f32_16x16x32_bf16(
            false, af[mt], false, bfr[nt], (short)0, acc[mt][nt], false, false);

    // ---- write next tile into the other buffer, flip ----
    if (more) {
      store_tileA(sA[cur ^ 1]);
      store_tileB(sB[cur ^ 1]);
    }
    __syncthreads();
    cur ^= 1;
  }

  // ---- epilogue: C/D layout: lane holds col (lane&15); rows hi*8 + r ----
#pragma unroll
  for (int mt = 0; mt < 4; mt++) {
#pragma unroll
    for (int nt = 0; nt < 2; nt++) {
#pragma unroll
      for (int r = 0; r < 8; r++) {
        int m = m0 + wm * 64 + mt * 16 + hi * 8 + r;
        int n = n0 + wn * 32 + nt * 16 + (lane & 15);
        float v = acc[mt][nt][r] * alpha;
        if (ROWSS) v = v * rowScale[m] + rowShift[m];
        if (COLB)  v += colBias[n];
        if (RELU)  v = fmaxf(v, 0.f);
        if (WF32)  Cf[(long)m * ldc + n] = v;
        if (WBF)   Cb[(long)m * ldc + n] = f2bf(v);
      }
    }
  }
}

// ---------------------------------------------------------------------------
__global__ void bn_prep_k(const float* __restrict__ bias, const float* __restrict__ gamma,
                          const float* __restrict__ beta, const float* __restrict__ mean,
                          const float* __restrict__ var,
                          float* __restrict__ scale, float* __restrict__ shift, int n) {
  int i = blockIdx.x * blockDim.x + threadIdx.x;
  if (i < n) {
    float s = gamma[i] * rsqrtf(var[i] + 1e-5f);
    scale[i] = s;
    shift[i] = beta[i] + (bias[i] - mean[i]) * s;
  }
}

__global__ void f2bf_k(const float* __restrict__ in, __bf16* __restrict__ out, long n) {
  long i = (long)blockIdx.x * blockDim.x + threadIdx.x;
  long st = (long)gridDim.x * blockDim.x;
  for (; i < n; i += st) out[i] = f2bf(in[i]);
}

// row softmax: grid (rows, z), block 256, f32 in -> bf16 out
__global__ __launch_bounds__(256)
void row_softmax_k(const float* __restrict__ in, __bf16* __restrict__ out,
                   int cols, long zIn, long zOut) {
  long rowb = (long)blockIdx.x * cols;
  in  += (long)blockIdx.y * zIn  + rowb;
  out += (long)blockIdx.y * zOut + rowb;
  __shared__ float red[256];
  int t = threadIdx.x;
  float mx = -1e30f;
  for (int c = t; c < cols; c += 256) mx = fmaxf(mx, in[c]);
  red[t] = mx; __syncthreads();
  for (int s = 128; s > 0; s >>= 1) { if (t < s) red[t] = fmaxf(red[t], red[t + s]); __syncthreads(); }
  mx = red[0]; __syncthreads();
  float sum = 0.f;
  for (int c = t; c < cols; c += 256) sum += expf(in[c] - mx);
  red[t] = sum; __syncthreads();
  for (int s = 128; s > 0; s >>= 1) { if (t < s) red[t] += red[t + s]; __syncthreads(); }
  float inv = 1.f / red[0];
  for (int c = t; c < cols; c += 256) out[c] = f2bf(expf(in[c] - mx) * inv);
}

// column softmax (softmax over rows): one thread per column
__global__ __launch_bounds__(256)
void col_softmax_k(const float* __restrict__ in, __bf16* __restrict__ out,
                   int rows, int cols, long zIn, long zOut) {
  int col = blockIdx.x * blockDim.x + threadIdx.x;
  if (col >= cols) return;
  in  += (long)blockIdx.y * zIn;
  out += (long)blockIdx.y * zOut;
  float mx = -1e30f;
  for (int r = 0; r < rows; r++) mx = fmaxf(mx, in[(long)r * cols + col]);
  float s = 0.f;
  for (int r = 0; r < rows; r++) s += expf(in[(long)r * cols + col] - mx);
  float inv = 1.f / s;
  for (int r = 0; r < rows; r++)
    out[(long)r * cols + col] = f2bf(expf(in[(long)r * cols + col] - mx) * inv);
}

// y = LN(O + X) * g + b   (grid: (S, B), block 256, E = 512)
template<bool WF32, bool WBF>
__global__ __launch_bounds__(256)
void resid_ln_k(const float* __restrict__ O, long oStride,
                const __bf16* __restrict__ X, long xStride,
                const float* __restrict__ g, const float* __restrict__ be,
                __bf16* __restrict__ outB, long obStride,
                float* __restrict__ outF, long ofStride, int E) {
  long row = blockIdx.x;
  int b = blockIdx.y, t = threadIdx.x;
  O += (long)b * oStride + row * E;
  X += (long)b * xStride + row * E;
  __shared__ float r1[256], r2[256];
  float s = 0.f, s2 = 0.f;
  for (int e = t; e < E; e += 256) { float y = O[e] + bf2f(X[e]); s += y; s2 += y * y; }
  r1[t] = s; r2[t] = s2; __syncthreads();
  for (int st = 128; st > 0; st >>= 1) {
    if (t < st) { r1[t] += r1[t + st]; r2[t] += r2[t + st]; }
    __syncthreads();
  }
  float mean = r1[0] / E;
  float var  = r2[0] / E - mean * mean;
  float inv  = rsqrtf(var + 1e-5f);
  if (WBF)  outB += (long)b * obStride + row * E;
  if (WF32) outF += (long)b * ofStride + row * E;
  for (int e = t; e < E; e += 256) {
    float y = O[e] + bf2f(X[e]);
    float r = (y - mean) * inv * g[e] + be[e];
    if (WBF)  outB[e] = f2bf(r);
    if (WF32) outF[e] = r;
  }
}

// ---------------------------------------------------------------------------
extern "C" void kernel_launch(void* const* d_in, const int* in_sizes, int n_in,
                              void* d_out, int out_size, void* d_ws, size_t ws_size,
                              hipStream_t stream) {
  (void)in_sizes; (void)n_in; (void)out_size; (void)ws_size;
  constexpr int Cc = 512, C2 = 1024, Ni = 1024, Np = 2048, Bn = 4, S3 = 3072, Hd = 128, NHh = 4;
  const float INV_SQRT_C = 0.044194173824159216f;  // 512^-0.5 (phi and attn scale)

  const float* img = (const float*)d_in[0];
  const float* pts = (const float*)d_in[1];
  const float* w1  = (const float*)d_in[2];
  const float* b1  = (const float*)d_in[3];
  const float* g1  = (const float*)d_in[4];
  const float* be1 = (const float*)d_in[5];
  const float* m1  = (const float*)d_in[6];
  const float* v1  = (const float*)d_in[7];
  const float* w2  = (const float*)d_in[8];
  const float* b2  = (const float*)d_in[9];
  const float* g2  = (const float*)d_in[10];
  const float* be2 = (const float*)d_in[11];
  const float* m2  = (const float*)d_in[12];
  const float* v2  = (const float*)d_in[13];
  const float* qw  = (const float*)d_in[14];
  const float* qbv = (const float*)d_in[15];
  const float* kw  = (const float*)d_in[16];
  const float* kbv = (const float*)d_in[17];
  const float* vw  = (const float*)d_in[18];
  const float* vbv = (const float*)d_in[19];
  const float* lng = (const float*)d_in[20];
  const float* lnb = (const float*)d_in[21];

  // ---- workspace layout ----
  char* base = (char*)d_ws;
  size_t off = 0;
  auto alloc = [&](size_t bytes) -> char* {
    char* p = base + off;
    off += (bytes + 255) & ~(size_t)255;
    return p;
  };
  float*  s1  = (float*)alloc((size_t)C2 * 4);
  float*  sh1 = (float*)alloc((size_t)C2 * 4);
  float*  s2  = (float*)alloc((size_t)Cc * 4);
  float*  sh2 = (float*)alloc((size_t)Cc * 4);
  __bf16* W1b = (__bf16*)alloc((size_t)C2 * Cc * 2);
  __bf16* W2b = (__bf16*)alloc((size_t)Cc * C2 * 2);
  __bf16* qwB = (__bf16*)alloc((size_t)3 * Cc * Cc * 2);
  __bf16* kwB = (__bf16*)alloc((size_t)3 * Cc * Cc * 2);
  __bf16* vwB = (__bf16*)alloc((size_t)3 * Cc * Cc * 2);
  __bf16* XpB = (__bf16*)alloc((size_t)Bn * Cc * Np * 2);
  __bf16* XiB = (__bf16*)alloc((size_t)Bn * Cc * Ni * 2);
  __bf16* pF  = (__bf16*)alloc((size_t)Bn * Cc * Np * 2);
  __bf16* iF  = (__bf16*)alloc((size_t)Bn * Cc * Ni * 2);
  __bf16* IT  = (__bf16*)alloc((size_t)Bn * Ni * Cc * 2);   // I_enh^T  [B,Ni,C]
  __bf16* PT  = (__bf16*)alloc((size_t)Bn * Np * Cc * 2);   // P_enh^T  [B,Np,C]
  __bf16* Jb  = (__bf16*)alloc((size_t)Bn * S3 * Cc * 2);   // joint    [B,3072,C]
  __bf16* Qb  = (__bf16*)alloc((size_t)Bn * S3 * Cc * 2);
  __bf16* Kb  = (__bf16*)alloc((size_t)Bn * S3 * Cc * 2);
  __bf16* Vb  = (__bf16*)alloc((size_t)Bn * S3 * Cc * 2);
  float*  Ob  = (float*)alloc((size_t)Bn * S3 * Cc * 4);
  // time-shared scratch (64 MB): MLP hidden | phi+softmaxes | scores
  char* scratch = alloc((size_t)64 * 1024 * 1024);
  __bf16* Yb      = (__bf16*)scratch;                                 // [B,1024,N]  16MB
  float*  phiF    = (float*)scratch;                                  // [B,Np,Ni]   32MB
  __bf16* phiP    = (__bf16*)(scratch + (size_t)32 * 1024 * 1024);    // 16MB
  __bf16* phiI    = (__bf16*)(scratch + (size_t)48 * 1024 * 1024);    // 16MB
  float*  scoresF = (float*)scratch;                                  // <=37.8MB
  __bf16* scoresB = (__bf16*)(scratch + (size_t)40 * 1024 * 1024);    // <=18.9MB

  // ---- BN folding + bf16 conversions ----
  bn_prep_k<<<dim3((C2 + 255) / 256), 256, 0, stream>>>(b1, g1, be1, m1, v1, s1, sh1, C2);
  bn_prep_k<<<dim3((Cc + 255) / 256), 256, 0, stream>>>(b2, g2, be2, m2, v2, s2, sh2, Cc);
  auto conv = [&](const float* src, __bf16* dst, long n) {
    long blocks = (n + 255) / 256; if (blocks > 8192) blocks = 8192;
    f2bf_k<<<dim3((unsigned)blocks), 256, 0, stream>>>(src, dst, n);
  };
  conv(w1, W1b, (long)C2 * Cc);
  conv(w2, W2b, (long)Cc * C2);
  conv(qw, qwB, 3L * Cc * Cc);
  conv(kw, kwB, 3L * Cc * Cc);
  conv(vw, vwB, 3L * Cc * Cc);
  conv(pts, XpB, (long)Bn * Cc * Np);
  conv(img, XiB, (long)Bn * Cc * Ni);

  // ---- shared MLP:  relu(BN(W2 x relu(BN(W1 x X)))) ----
  auto run_mlp = [&](const __bf16* X, int N, __bf16* outFeat) {
    gemm_bf16<false, false, true, false, true, false, true>
        <<<dim3(N / 128, C2 / 128, Bn), 256, 0, stream>>>(
        W1b, X, nullptr, Yb, C2, N, Cc, Cc, N, N,
        0L, (long)Cc * N, (long)C2 * N, s1, sh1, nullptr, 1.f);
    gemm_bf16<false, false, true, false, true, false, true>
        <<<dim3(N / 128, Cc / 128, Bn), 256, 0, stream>>>(
        W2b, Yb, nullptr, outFeat, Cc, N, C2, C2, N, N,
        0L, (long)C2 * N, (long)Cc * N, s2, sh2, nullptr, 1.f);
  };
  run_mlp(XpB, Np, pF);
  run_mlp(XiB, Ni, iF);

  // ---- cross attention: phi = pF^T iF / sqrt(C) ----
  gemm_bf16<true, false, false, false, false, true, false>
      <<<dim3(Ni / 128, Np / 128, Bn), 256, 0, stream>>>(
      pF, iF, phiF, nullptr, Np, Ni, Cc, Np, Ni, Ni,
      (long)Cc * Np, (long)Cc * Ni, (long)Np * Ni, nullptr, nullptr, nullptr, INV_SQRT_C);
  col_softmax_k<<<dim3(Ni / 256, Bn), 256, 0, stream>>>(
      phiF, phiP, Np, Ni, (long)Np * Ni, (long)Np * Ni);
  row_softmax_k<<<dim3(Np, Bn), 256, 0, stream>>>(
      phiF, phiI, Ni, (long)Np * Ni, (long)Np * Ni);
  // I_enh^T[i,c] = sum_p phiP[p,i] pF[c,p]
  gemm_bf16<true, true, false, false, false, false, true>
      <<<dim3(Cc / 128, Ni / 128, Bn), 256, 0, stream>>>(
      phiP, pF, nullptr, IT, Ni, Cc, Np, Ni, Np, Cc,
      (long)Np * Ni, (long)Cc * Np, (long)Ni * Cc, nullptr, nullptr, nullptr, 1.f);
  // P_enh^T[p,c] = sum_i phiI[p,i] iF[c,i]
  gemm_bf16<false, true, false, false, false, false, true>
      <<<dim3(Cc / 128, Np / 128, Bn), 256, 0, stream>>>(
      phiI, iF, nullptr, PT, Np, Cc, Ni, Ni, Ni, Cc,
      (long)Np * Ni, (long)Cc * Ni, (long)Np * Cc, nullptr, nullptr, nullptr, 1.f);

  // ---- self-attention block ----
  auto run_attn = [&](const __bf16* X, long xsb, int S, int j,
                      __bf16* outB, long obStride, float* outF) {
    const __bf16* pw[3] = {qwB, kwB, vwB};
    const float*  pb[3] = {qbv, kbv, vbv};
    __bf16* dst[3] = {Qb, Kb, Vb};
    for (int t = 0; t < 3; t++) {
      gemm_bf16<false, true, false, true, false, false, true>
          <<<dim3(Cc / 128, S / 128, Bn), 256, 0, stream>>>(
          X, pw[t] + (size_t)j * Cc * Cc, nullptr, dst[t], S, Cc, Cc, Cc, Cc, Cc,
          xsb, 0L, (long)S * Cc, nullptr, nullptr, pb[t] + (size_t)j * Cc, 1.f);
    }
    for (int b = 0; b < Bn; b++) {
      for (int h = 0; h < NHh; h++) {
        const __bf16* Qp = Qb + (long)b * S * Cc + h * Hd;
        const __bf16* Kp = Kb + (long)b * S * Cc + h * Hd;
        const __bf16* Vp = Vb + (long)b * S * Cc + h * Hd;
        float* Op = Ob + (long)b * S * Cc + h * Hd;
        gemm_bf16<false, true, false, false, false, true, false>
            <<<dim3(S / 128, S / 128, 1), 256, 0, stream>>>(
            Qp, Kp, scoresF, nullptr, S, S, Hd, Cc, Cc, S,
            0L, 0L, 0L, nullptr, nullptr, nullptr, INV_SQRT_C);
        row_softmax_k<<<dim3(S, 1), 256, 0, stream>>>(scoresF, scoresB, S, 0L, 0L);
        gemm_bf16<false, false, false, false, false, true, false>
            <<<dim3(1, S / 128, 1), 256, 0, stream>>>(
            scoresB, Vp, Op, nullptr, S, Hd, S, S, Cc, Cc,
            0L, 0L, 0L, nullptr, nullptr, nullptr, 1.f);
      }
    }
    if (outF) {
      resid_ln_k<true, false><<<dim3(S, Bn), 256, 0, stream>>>(
          Ob, (long)S * Cc, X, xsb, lng + (size_t)j * Cc, lnb + (size_t)j * Cc,
          nullptr, 0L, outF, (long)S * Cc, Cc);
    } else {
      resid_ln_k<false, true><<<dim3(S, Bn), 256, 0, stream>>>(
          Ob, (long)S * Cc, X, xsb, lng + (size_t)j * Cc, lnb + (size_t)j * Cc,
          outB, obStride, nullptr, 0L, Cc);
    }
  };

  // block 0: I = attn(I_enh^T) -> joint rows [2048,3072)
  run_attn(IT, (long)Ni * Cc, Ni, 0, Jb + (long)Np * Cc, (long)S3 * Cc, nullptr);
  // block 1: P = attn(P_enh^T) -> joint rows [0,2048)
  run_attn(PT, (long)Np * Cc, Np, 1, Jb, (long)S3 * Cc, nullptr);
  // block 2: out = attn(joint) -> d_out (f32)
  run_attn(Jb, (long)S3 * Cc, S3, 2, nullptr, 0L, (float*)d_out);
}
